// WeightedSumGraphRepresentation_83056077570616
// MI455X (gfx1250) — matrix operand order repaired and checked by reading, not compiled, not run on hardware
//
#include <hip/hip_runtime.h>

// ---------------------------------------------------------------------------
// MI455X (gfx1250) fused graph-attention readout.
// All GEMMs use v_wmma_f32_16x16x32_bf16 (wave32, f32 accumulate).
// Each wave register-blocks M=32 (two 16-row tiles) so every B fragment
// (2x global_load_b128 from the L2-resident repacked weights) feeds 2 WMMAs.
// ---------------------------------------------------------------------------

typedef __attribute__((ext_vector_type(16))) __bf16 v16bf;   // A/B fragment (8 VGPRs)
typedef __attribute__((ext_vector_type(8)))  __bf16 v8bf;
typedef __attribute__((ext_vector_type(8)))  float  v8f;     // C/D fragment (8 VGPRs)

#define WMMA_BF16(a, b, c) \
  __builtin_amdgcn_wmma_f32_16x16x32_bf16(false, (a), false, (b), (short)0, (c), false, false)

// Fragment-index bases inside the prepped bf16 weight buffer (1 KB / fragment).
#define FR_W1S 0     // 8 kt x 8 nt   (256x128)
#define FR_W1T 64    // 8 kt x 8 nt   (256x128)
#define FR_W2T 128   // 4 kt x 16 nt  (128x256)
#define FR_W2S 192   // 4 kt x 1 nt   (128x8, zero-padded to 16 cols)
#define NFRAG  196

__device__ __forceinline__ int lower_bound_i32(const int* a, int n, int v) {
  int lo = 0, hi = n;
  while (lo < hi) { int mid = (lo + hi) >> 1; if (a[mid] < v) lo = mid + 1; else hi = mid; }
  return lo;
}

// A-matrix 16x32 bf16 fragment from a global f32 row (ISA 7.12.2 16-bit A layout):
// lane L holds row (L&15); halves h<8 -> K = half*8+h, h>=8 -> K = half*8+8+h.
__device__ __forceinline__ v16bf load_a_global(const float* rowp, int k0, int half) {
  const float* p = rowp + k0 + half * 8;
  float4 a0 = *(const float4*)(p + 0);
  float4 a1 = *(const float4*)(p + 4);
  float4 b0 = *(const float4*)(p + 16);
  float4 b1 = *(const float4*)(p + 20);
  v16bf f;
  f[0]  = (__bf16)a0.x; f[1]  = (__bf16)a0.y; f[2]  = (__bf16)a0.z; f[3]  = (__bf16)a0.w;
  f[4]  = (__bf16)a1.x; f[5]  = (__bf16)a1.y; f[6]  = (__bf16)a1.z; f[7]  = (__bf16)a1.w;
  f[8]  = (__bf16)b0.x; f[9]  = (__bf16)b0.y; f[10] = (__bf16)b0.z; f[11] = (__bf16)b0.w;
  f[12] = (__bf16)b1.x; f[13] = (__bf16)b1.y; f[14] = (__bf16)b1.z; f[15] = (__bf16)b1.w;
  return f;
}

// Same A layout, sourced from a row-major bf16 LDS row (two ds_load_b128 per lane).
__device__ __forceinline__ v16bf load_a_lds(const __bf16* rowp, int k0, int half) {
  const v8bf* p0 = (const v8bf*)(rowp + k0 + half * 8);
  const v8bf* p1 = (const v8bf*)(rowp + k0 + half * 8 + 16);
  v8bf lo = *p0, hi = *p1;
  return __builtin_shufflevector(lo, hi, 0,1,2,3,4,5,6,7,8,9,10,11,12,13,14,15);
}

// B fragment: 32 bytes contiguous per lane from the prepped buffer (L2 resident).
__device__ __forceinline__ v16bf load_b_frag(const __bf16* wf, int frag, int lane) {
  return *(const v16bf*)(wf + ((size_t)frag << 9) + (lane << 4));
}

// ---------------------------------------------------------------------------
// Kernel 0: repack f32 weights into bf16 B-fragment layout.
// B-matrix 32x16 layout (ISA 7.12.5 dense-B): lane L = col (L&15),
// lanes 0-15 hold K = kt*32 + 0..7,16..23, lanes 16-31 hold K = kt*32 + 8..15,24..31.
// ---------------------------------------------------------------------------
__global__ void k_prep(const float* __restrict__ w1s, const float* __restrict__ w2s,
                       const float* __restrict__ w1t, const float* __restrict__ w2t,
                       __bf16* __restrict__ wfrags)
{
  int gid  = blockIdx.x * blockDim.x + threadIdx.x;   // one thread per (frag, lane)
  int frag = gid >> 5;
  int lane = gid & 31;
  if (frag >= NFRAG) return;
  int half = lane >> 4, n0 = lane & 15;

  const float* W; int kt, nt, ld, nvalid;
  if (frag < FR_W1T)      { W = w1s; int f = frag - FR_W1S; kt = f >> 3; nt = f & 7;  ld = 128; nvalid = 128; }
  else if (frag < FR_W2T) { W = w1t; int f = frag - FR_W1T; kt = f >> 3; nt = f & 7;  ld = 128; nvalid = 128; }
  else if (frag < FR_W2S) { W = w2t; int f = frag - FR_W2T; kt = f >> 4; nt = f & 15; ld = 256; nvalid = 256; }
  else                    { W = w2s; int f = frag - FR_W2S; kt = f;      nt = 0;      ld = 8;   nvalid = 8;   }

  int n = nt * 16 + n0;
  v16bf out;
#pragma unroll
  for (int h = 0; h < 16; ++h) {
    int k = kt * 32 + (h < 8 ? half * 8 + h : 16 + half * 8 + (h - 8));
    float v = (n < nvalid) ? W[k * ld + n] : 0.0f;
    out[h] = (__bf16)v;
  }
  *(v16bf*)(wfrags + ((size_t)frag << 9) + (lane << 4)) = out;
}

// ---------------------------------------------------------------------------
// Kernel 1: scores[V,8] = relu(E @ W1s) @ W2s.  One wave per 32-node tile pair.
// ---------------------------------------------------------------------------
__global__ void __launch_bounds__(128) k_scores(const float* __restrict__ E,
                                                const __bf16* __restrict__ wfrags,
                                                float* __restrict__ scores, int V)
{
  int wave = threadIdx.x >> 5, lane = threadIdx.x & 31;
  int half = lane >> 4, l15 = lane & 15;
  int tiles  = (V + 15) >> 4;
  int npairs = (tiles + 1) >> 1;
  int tp = blockIdx.x * 4 + wave;
  __shared__ __align__(32) __bf16 hid[4][2][16][128];
  if (tp >= npairs) return;

  int base0 = tp * 32;
  if (tp + 4 < npairs)
    __builtin_prefetch(E + (size_t)(base0 + 128 + l15) * 256, 0, 1);

  v16bf a[2][8];
#pragma unroll
  for (int sb = 0; sb < 2; ++sb) {
    int row = base0 + sb * 16 + l15; if (row >= V) row = V - 1;
    const float* erow = E + (size_t)row * 256;
#pragma unroll
    for (int kt = 0; kt < 8; ++kt) a[sb][kt] = load_a_global(erow, kt * 32, half);
  }

#pragma unroll
  for (int nt = 0; nt < 8; ++nt) {
    v8f c0 = {0.f, 0.f, 0.f, 0.f, 0.f, 0.f, 0.f, 0.f};
    v8f c1 = {0.f, 0.f, 0.f, 0.f, 0.f, 0.f, 0.f, 0.f};
#pragma unroll
    for (int kt = 0; kt < 8; ++kt) {
      v16bf b = load_b_frag(wfrags, FR_W1S + kt * 8 + nt, lane);
      c0 = WMMA_BF16(a[0][kt], b, c0);     // one B fragment feeds both M tiles
      c1 = WMMA_BF16(a[1][kt], b, c1);
    }
    // C layout: lane = col (l15), VGPR r = row r + 8*half. ReLU, transpose via LDS.
#pragma unroll
    for (int r = 0; r < 8; ++r) {
      hid[wave][0][r + 8 * half][nt * 16 + l15] = (__bf16)fmaxf(c0[r], 0.f);
      hid[wave][1][r + 8 * half][nt * 16 + l15] = (__bf16)fmaxf(c1[r], 0.f);
    }
  }
  asm volatile("s_wait_dscnt 0" ::: "memory");

  v16bf ha[2][4];
#pragma unroll
  for (int sb = 0; sb < 2; ++sb)
#pragma unroll
    for (int kt = 0; kt < 4; ++kt)
      ha[sb][kt] = load_a_lds(&hid[wave][sb][l15][0], kt * 32, half);

  v8f sc0 = {0.f, 0.f, 0.f, 0.f, 0.f, 0.f, 0.f, 0.f};
  v8f sc1 = {0.f, 0.f, 0.f, 0.f, 0.f, 0.f, 0.f, 0.f};
#pragma unroll
  for (int kt = 0; kt < 4; ++kt) {
    v16bf b = load_b_frag(wfrags, FR_W2S + kt, lane);
    sc0 = WMMA_BF16(ha[0][kt], b, sc0);
    sc1 = WMMA_BF16(ha[1][kt], b, sc1);
  }
  if (l15 < 8) {                 // only 8 head columns are valid
#pragma unroll
    for (int r = 0; r < 8; ++r) {
      int rr0 = base0 + r + 8 * half;
      if (rr0 < V) scores[(size_t)rr0 * 8 + l15] = sc0[r];
      int rr1 = base0 + 16 + r + 8 * half;
      if (rr1 < V) scores[(size_t)rr1 * 8 + l15] = sc1[r];
    }
  }
}

// ---------------------------------------------------------------------------
// Kernel 2: per-graph, per-head softmax stats (max, sum-exp) over the sorted map.
// ---------------------------------------------------------------------------
__global__ void __launch_bounds__(256) k_stats(const float* __restrict__ scores,
                                               const int* __restrict__ map,
                                               float* __restrict__ gmax,
                                               float* __restrict__ gden, int V)
{
  int g = blockIdx.x;
  int start = lower_bound_i32(map, V, g);
  int end   = lower_bound_i32(map, V, g + 1);
  int h = threadIdx.x & 7, sub = threadIdx.x >> 3;   // 8 heads x 32 sub-lanes

  __shared__ float red[8][32];
  __shared__ float bm[8];

  float m = -3.4e38f;
  for (int i = start + sub; i < end; i += 32) m = fmaxf(m, scores[(size_t)i * 8 + h]);
  red[h][sub] = m;
  __syncthreads();
  if (threadIdx.x < 8) {
    float mm = red[threadIdx.x][0];
    for (int j = 1; j < 32; ++j) mm = fmaxf(mm, red[threadIdx.x][j]);
    if (mm < -1e37f) mm = 0.f;               // jnp.where(isfinite, max, 0)
    bm[threadIdx.x] = mm;
  }
  __syncthreads();
  float gm = bm[h];
  float s = 0.f;
  for (int i = start + sub; i < end; i += 32) s += __expf(scores[(size_t)i * 8 + h] - gm);
  __syncthreads();
  red[h][sub] = s;
  __syncthreads();
  if (threadIdx.x < 8) {
    float ss = 0.f;
    for (int j = 0; j < 32; ++j) ss += red[threadIdx.x][j];
    gmax[(size_t)g * 8 + threadIdx.x] = bm[threadIdx.x];
    gden[(size_t)g * 8 + threadIdx.x] = ss;
  }
}

// ---------------------------------------------------------------------------
// Kernel 3: out[g,:] = sum_nodes softmax_w * relu(relu(E@W1t)@W2t), reshaped heads.
// One block (4 waves) per graph; each wave processes 32-node tile pairs.
// Sorted map -> contiguous node range, no global atomics.
// ---------------------------------------------------------------------------
__global__ void __launch_bounds__(128) k_accum(const float* __restrict__ E,
                                               const __bf16* __restrict__ wfrags,
                                               const float* __restrict__ scores,
                                               const float* __restrict__ gmax,
                                               const float* __restrict__ gden,
                                               const int* __restrict__ map,
                                               float* __restrict__ out, int V)
{
  int g = blockIdx.x;
  int tid = threadIdx.x;
  int wave = tid >> 5, lane = tid & 31, half = lane >> 4, l15 = lane & 15;

  __shared__ float acc[256];
  __shared__ __align__(32) __bf16 hid[4][2][16][128];
  __shared__ float wbuf[4][2][16][8];

  for (int i = tid; i < 256; i += 128) acc[i] = 0.f;
  __syncthreads();

  int start = lower_bound_i32(map, V, g);
  int end   = lower_bound_i32(map, V, g + 1);
  int ntiles = (end - start + 15) >> 4;
  int npairs = (ntiles + 1) >> 1;

  for (int tp = wave; tp < npairs; tp += 4) {
    int nb0 = start + tp * 32;

    // Softmax weights for both 16-node tiles -> LDS (4 entries per lane per tile).
    v16bf a[2][8];
#pragma unroll
    for (int sb = 0; sb < 2; ++sb) {
      int nb = nb0 + sb * 16;
#pragma unroll
      for (int j = 0; j < 4; ++j) {
        int idx = lane * 4 + j;
        int nrow = idx >> 3, hh = idx & 7;
        int grow = nb + nrow;
        float w = 0.f;
        if (grow < end) {
          float mx = gmax[(size_t)g * 8 + hh];
          float dn = gden[(size_t)g * 8 + hh];
          w = __expf(scores[(size_t)grow * 8 + hh] - mx) / dn;
        }
        wbuf[wave][sb][nrow][hh] = w;
      }
      int row = nb + l15;
      int rowc = (row < end) ? row : (end - 1);        // pad rows get weight 0
      const float* erow = E + (size_t)rowc * 256;
#pragma unroll
      for (int kt = 0; kt < 8; ++kt) a[sb][kt] = load_a_global(erow, kt * 32, half);
    }

#pragma unroll
    for (int nt = 0; nt < 8; ++nt) {
      v8f c0 = {0.f, 0.f, 0.f, 0.f, 0.f, 0.f, 0.f, 0.f};
      v8f c1 = {0.f, 0.f, 0.f, 0.f, 0.f, 0.f, 0.f, 0.f};
#pragma unroll
      for (int kt = 0; kt < 8; ++kt) {
        v16bf b = load_b_frag(wfrags, FR_W1T + kt * 8 + nt, lane);
        c0 = WMMA_BF16(a[0][kt], b, c0);
        c1 = WMMA_BF16(a[1][kt], b, c1);
      }
#pragma unroll
      for (int r = 0; r < 8; ++r) {
        hid[wave][0][r + 8 * half][nt * 16 + l15] = (__bf16)fmaxf(c0[r], 0.f);
        hid[wave][1][r + 8 * half][nt * 16 + l15] = (__bf16)fmaxf(c1[r], 0.f);
      }
    }
    asm volatile("s_wait_dscnt 0" ::: "memory");

    v16bf ha[2][4];
#pragma unroll
    for (int sb = 0; sb < 2; ++sb)
#pragma unroll
      for (int kt = 0; kt < 4; ++kt)
        ha[sb][kt] = load_a_lds(&hid[wave][sb][l15][0], kt * 32, half);

#pragma unroll
    for (int ct = 0; ct < 16; ++ct) {               // 16 col-tiles of the [*,256] reprs
      v8f c0 = {0.f, 0.f, 0.f, 0.f, 0.f, 0.f, 0.f, 0.f};
      v8f c1 = {0.f, 0.f, 0.f, 0.f, 0.f, 0.f, 0.f, 0.f};
#pragma unroll
      for (int kt = 0; kt < 4; ++kt) {
        v16bf b = load_b_frag(wfrags, FR_W2T + kt * 16 + ct, lane);
        c0 = WMMA_BF16(ha[0][kt], b, c0);
        c1 = WMMA_BF16(ha[1][kt], b, c1);
      }
      int hh = ct >> 1;                             // head = column / 32
      float partial = 0.f;
#pragma unroll
      for (int r = 0; r < 8; ++r) {
        partial += wbuf[wave][0][r + 8 * half][hh] * fmaxf(c0[r], 0.f);
        partial += wbuf[wave][1][r + 8 * half][hh] * fmaxf(c1[r], 0.f);
      }
      atomicAdd(&acc[ct * 16 + l15], partial);      // one ds_add_f32 per pair
    }
  }
  __syncthreads();
  for (int i = tid; i < 256; i += 128) out[(size_t)g * 256 + i] = acc[i];
}

// ---------------------------------------------------------------------------
extern "C" void kernel_launch(void* const* d_in, const int* in_sizes, int n_in,
                              void* d_out, int out_size, void* d_ws, size_t ws_size,
                              hipStream_t stream)
{
  const float* E    = (const float*)d_in[0];   // [V,256]
  const int*   map  = (const int*)d_in[1];     // [V] sorted graph ids
  const float* w1s  = (const float*)d_in[3];   // [256,128]
  const float* w2s  = (const float*)d_in[4];   // [128,8]
  const float* w1t  = (const float*)d_in[5];   // [256,128]
  const float* w2t  = (const float*)d_in[6];   // [128,256]
  float* out = (float*)d_out;

  const int V = in_sizes[0] / 256;
  const int G = out_size / 256;

  // Workspace layout (all offsets 256B aligned):
  //   [0, 200704)             bf16 weight fragments (196 x 1KB)
  //   [200704, +V*32)         scores f32 [V,8]
  //   then gmax [G,8], gden [G,8]
  char* ws = (char*)d_ws;
  __bf16* wfrags = (__bf16*)ws;
  float*  scores = (float*)(ws + (size_t)NFRAG * 1024);
  float*  gmax   = (float*)(ws + (size_t)NFRAG * 1024 + (size_t)V * 32);
  float*  gden   = gmax + (size_t)G * 8;

  k_prep<<<(NFRAG * 32 + 255) / 256, 256, 0, stream>>>(w1s, w2s, w1t, w2t, wfrags);

  int tiles  = (V + 15) / 16;
  int npairs = (tiles + 1) / 2;
  k_scores<<<(npairs + 3) / 4, 128, 0, stream>>>(E, wfrags, scores, V);

  k_stats<<<G, 256, 0, stream>>>(scores, map, gmax, gden, V);

  k_accum<<<G, 128, 0, stream>>>(E, wfrags, scores, gmax, gden, map, out, V);
}
